// SDE_36146444763416
// MI455X (gfx1250) — matrix operand order repaired and checked by reading, not compiled, not run on hardware
//
#include <hip/hip_runtime.h>

// ---------------------------------------------------------------------------
// SDE embedding, single-pass online-softmax (flash) over the latent basis,
// bf16 WMMA, Mtile=32 for 2x B-operand reuse (L2 traffic ~5GB vs 10GB).
//   M = 8192 token rows, D = 512, L = 10000, 20 chars/token.
// ---------------------------------------------------------------------------

#define DIM      512
#define LATENT   10000
#define ROWS     8192
#define NCHAR    20
#define NTILES   625            // LATENT / 16
#define NCHUNK   79             // ceil(NTILES / 8) chunks of 128 latent rows
#define LT_STRIDE 10112         // padded transposed-latent row length (zero tail)
#define AST      520            // A_lds stride in bf16 (4-dword bank skew)
#define PST      136            // P_lds stride in bf16 (4-dword bank skew)
#define TT       64             // transpose tile
#define MT       32             // rows per block (2 x 16 M-subtiles)

typedef __attribute__((ext_vector_type(16))) __bf16       v16bf;
typedef __attribute__((ext_vector_type(8)))  float        v8f;
typedef __attribute__((ext_vector_type(8)))  unsigned int v8u;

union AFrag { uint4 q[2]; v16bf v; };   // two contiguous 16B halves
union BFrag { v8u   u;    v16bf v; };   // one contiguous 32B chunk
union QW    { unsigned short h[8]; uint4 q; };

__device__ __forceinline__ unsigned short f32_to_bf16(float f) {
  unsigned u = __float_as_uint(f);
  u += 0x7FFFu + ((u >> 16) & 1u);      // round-to-nearest-even
  return (unsigned short)(u >> 16);
}

#define WMMA_BF16(A, B, C) \
  __builtin_amdgcn_wmma_f32_16x16x32_bf16(false, (A), false, (B), (short)0, (C), false, false)

// ---- latent f32 -> bf16, row major (coalesced) ----------------------------
__global__ void k_convert(const float* __restrict__ latent,
                          unsigned short* __restrict__ latB) {
  int idx = blockIdx.x * 256 + threadIdx.x;
  if (idx < LATENT * DIM) latB[idx] = f32_to_bf16(latent[idx]);
}

// ---- LDS-tiled 64x64 transpose: latent(L x D) f32 -> latT(D x LT_STRIDE) --
__global__ void k_transpose(const float* __restrict__ latent,
                            unsigned short* __restrict__ latT) {
  __shared__ unsigned short tile[TT][TT + 8];
  const int lb  = blockIdx.x * TT;          // 158 tiles cover LT_STRIDE exactly
  const int db  = blockIdx.y * TT;
  const int tid = threadIdx.x;              // 256 threads
  #pragma unroll
  for (int i = 0; i < 4; ++i) {             // read 64x64 f32, coalesced float4
    int idx = tid + i * 256;
    int r = idx >> 4, cf = idx & 15;
    int l = lb + r;
    float4 v = make_float4(0.f, 0.f, 0.f, 0.f);
    if (l < LATENT) v = ((const float4*)(latent + (size_t)l * DIM + db))[cf];
    tile[r][cf * 4 + 0] = f32_to_bf16(v.x);
    tile[r][cf * 4 + 1] = f32_to_bf16(v.y);
    tile[r][cf * 4 + 2] = f32_to_bf16(v.z);
    tile[r][cf * 4 + 3] = f32_to_bf16(v.w);
  }
  __syncthreads();
  #pragma unroll
  for (int i = 0; i < 2; ++i) {             // write 64x64 bf16, coalesced uint4
    int idx = tid + i * 256;
    int r = idx >> 3, cq = idx & 7;
    QW tw;
    #pragma unroll
    for (int j = 0; j < 8; ++j) tw.h[j] = tile[cq * 8 + j][r];
    *(uint4*)(&latT[(size_t)(db + r) * LT_STRIDE + lb + cq * 8]) = tw.q;
  }
}

// ---- char gather + masked sum + tanh --------------------------------------
__global__ void k_lang(const int* __restrict__ x, const float* __restrict__ ctab,
                       float* __restrict__ langF, unsigned short* __restrict__ langB) {
  int row = blockIdx.x;
  int c0 = threadIdx.x, c1 = threadIdx.x + 256;
  const int* xr = x + row * NCHAR;
  float a0 = 0.f, a1 = 0.f;
  #pragma unroll
  for (int i = 0; i < NCHAR; ++i) {
    int c = xr[i];
    if (c != 1) {                           // PAD == 1
      const float* tr = ctab + (size_t)c * DIM;
      a0 += tr[c0]; a1 += tr[c1];
    }
  }
  a0 = tanhf(a0); a1 = tanhf(a1);
  size_t base = (size_t)row * DIM;
  langF[base + c0] = a0;                 langF[base + c1] = a1;
  langB[base + c0] = f32_to_bf16(a0);    langB[base + c1] = f32_to_bf16(a1);
}

// ---- fused single-pass: scores + online softmax + O-GEMM + epilogue -------
__global__ __launch_bounds__(256)
void k_sde(const int* __restrict__ x, const float* __restrict__ ctab,
           const float* __restrict__ langF, const unsigned short* __restrict__ langB,
           const unsigned short* __restrict__ latB, const unsigned short* __restrict__ latT,
           float* __restrict__ out) {
  __shared__ __align__(16) unsigned short A_lds[MT * AST];
  __shared__ __align__(16) unsigned short P_lds[MT * PST];
  __shared__ float redA[8][MT];             // per-wave row stats
  __shared__ float redB[MT];                // merged per-row value

  const int tid  = threadIdx.x;
  const int w    = tid >> 5, lane = tid & 31;
  const int l16  = lane & 15, hi = lane >> 4;
  const int mbase = blockIdx.x * MT;

  // ---- stage 32 x 512 bf16 A tile into LDS (bank-skewed stride) ----
  {
    const uint4* src = (const uint4*)(langB + (size_t)mbase * DIM);
    #pragma unroll
    for (int i = 0; i < 8; ++i) {
      int idx = tid + i * 256;              // 2048 x 16B
      int r = idx >> 6, cq = idx & 63;
      uint4 v = src[r * 64 + cq];
      *(uint4*)(&A_lds[r * AST + cq * 8]) = v;
    }
  }
  __syncthreads();

  const char* Ab0 = (const char*)A_lds + l16 * (AST * 2) + hi * 32;
  const char* Ab1 = Ab0 + 16 * (AST * 2);
  const char* Pb0 = (const char*)P_lds + l16 * (PST * 2) + hi * 32;
  const char* Pb1 = Pb0 + 16 * (PST * 2);
  const char* latTc = (const char*)latT;
  const int dbase = w * 64;

  float m0[8], s0[8], m1[8], s1[8];
  #pragma unroll
  for (int r = 0; r < 8; ++r) {
    m0[r] = -3.0e38f; s0[r] = 0.f; m1[r] = -3.0e38f; s1[r] = 0.f;
  }
  v8f oa0 = {}, oa1 = {}, oa2 = {}, oa3 = {};   // Msub0 rows 0..15
  v8f ob0 = {}, ob1 = {}, ob2 = {}, ob3 = {};   // Msub1 rows 16..31

  int firA[8], firB[8];
  #pragma unroll
  for (int r = 0; r < 8; ++r) {
    firA[r] = x[(size_t)(mbase +      r + 8 * hi) * NCHAR];
    firB[r] = x[(size_t)(mbase + 16 + r + 8 * hi) * NCHAR];
  }

  for (int c = 0; c < NCHUNK; ++c) {
    const int t = c * 8 + w;                // this wave's 16-wide latent tile
    v8f acc0 = {}, acc1 = {};
    float tm0[8], tm1[8];
    if (t < NTILES) {
      const char* Bb = (const char*)latB + ((size_t)(t * 16 + l16)) * (DIM * 2) + hi * 32;
      __builtin_prefetch(Bb + 8 * 16 * DIM * 2, 0, 1);   // next chunk's rows
      #pragma unroll
      for (int ks = 0; ks < 16; ++ks) {
        BFrag b; b.u = *(const v8u*)(Bb + ks * 64);      // shared by both Msubs
        AFrag a0; a0.q[0] = *(const uint4*)(Ab0 + ks * 64);
                  a0.q[1] = *(const uint4*)(Ab0 + ks * 64 + 16);
        AFrag a1; a1.q[0] = *(const uint4*)(Ab1 + ks * 64);
                  a1.q[1] = *(const uint4*)(Ab1 + ks * 64 + 16);
        acc0 = WMMA_BF16(a0.v, b.v, acc0);
        acc1 = WMMA_BF16(a1.v, b.v, acc1);
      }
      #pragma unroll
      for (int r = 0; r < 8; ++r) { tm0[r] = acc0[r]; tm1[r] = acc1[r]; }
    } else {
      #pragma unroll
      for (int r = 0; r < 8; ++r) { tm0[r] = -3.0e38f; tm1[r] = -3.0e38f; }
    }
    // intra-wave row max over this wave's 16 columns
    #pragma unroll
    for (int r = 0; r < 8; ++r) {
      #pragma unroll
      for (int msk = 1; msk < 16; msk <<= 1) {
        tm0[r] = fmaxf(tm0[r], __shfl_xor(tm0[r], msk, 16));
        tm1[r] = fmaxf(tm1[r], __shfl_xor(tm1[r], msk, 16));
      }
    }
    if (l16 == 0) {
      #pragma unroll
      for (int r = 0; r < 8; ++r) {
        redA[w][r + 8 * hi]      = tm0[r];
        redA[w][16 + r + 8 * hi] = tm1[r];
      }
    }
    __syncthreads();
    if (tid < MT) {                         // chunk-wide row max across waves
      float cm = redA[0][tid];
      #pragma unroll
      for (int wv = 1; wv < 8; ++wv) cm = fmaxf(cm, redA[wv][tid]);
      redB[tid] = cm;
    }
    __syncthreads();

    // online update: rescale running sums + accumulators, emit P (bf16) to LDS
    #pragma unroll
    for (int r = 0; r < 8; ++r) {
      float mn0   = fmaxf(m0[r], redB[r + 8 * hi]);
      float corr0 = __expf(m0[r] - mn0);
      m0[r] = mn0;  s0[r] *= corr0;
      oa0[r] *= corr0; oa1[r] *= corr0; oa2[r] *= corr0; oa3[r] *= corr0;
      float mn1   = fmaxf(m1[r], redB[16 + r + 8 * hi]);
      float corr1 = __expf(m1[r] - mn1);
      m1[r] = mn1;  s1[r] *= corr1;
      ob0[r] *= corr1; ob1[r] *= corr1; ob2[r] *= corr1; ob3[r] *= corr1;
      if (t < NTILES) {
        float p0 = __expf(acc0[r] - mn0);
        float p1 = __expf(acc1[r] - mn1);
        s0[r] += p0; s1[r] += p1;
        P_lds[(r + 8 * hi) * PST + w * 16 + l16]        = f32_to_bf16(p0);
        P_lds[(16 + r + 8 * hi) * PST + w * 16 + l16]   = f32_to_bf16(p1);
      } else {
        P_lds[(r + 8 * hi) * PST + w * 16 + l16]        = 0;
        P_lds[(16 + r + 8 * hi) * PST + w * 16 + l16]   = 0;
      }
    }
    __syncthreads();                        // P ready

    // O += P(32x128) x latentT(128 x this wave's 64 cols); B reused by 2 Msubs
    #pragma unroll
    for (int ks = 0; ks < 4; ++ks) {
      AFrag a0; a0.q[0] = *(const uint4*)(Pb0 + ks * 64);
                a0.q[1] = *(const uint4*)(Pb0 + ks * 64 + 16);
      AFrag a1; a1.q[0] = *(const uint4*)(Pb1 + ks * 64);
                a1.q[1] = *(const uint4*)(Pb1 + ks * 64 + 16);
      const size_t kb = (size_t)c * 256 + ks * 64 + hi * 32;
      BFrag b0, b1, b2, b3;                 // preload all 4 for overlap
      b0.u = *(const v8u*)(latTc + (size_t)(dbase +  0 + l16) * (LT_STRIDE * 2) + kb);
      b1.u = *(const v8u*)(latTc + (size_t)(dbase + 16 + l16) * (LT_STRIDE * 2) + kb);
      b2.u = *(const v8u*)(latTc + (size_t)(dbase + 32 + l16) * (LT_STRIDE * 2) + kb);
      b3.u = *(const v8u*)(latTc + (size_t)(dbase + 48 + l16) * (LT_STRIDE * 2) + kb);
      oa0 = WMMA_BF16(a0.v, b0.v, oa0);  ob0 = WMMA_BF16(a1.v, b0.v, ob0);
      oa1 = WMMA_BF16(a0.v, b1.v, oa1);  ob1 = WMMA_BF16(a1.v, b1.v, ob1);
      oa2 = WMMA_BF16(a0.v, b2.v, oa2);  ob2 = WMMA_BF16(a1.v, b2.v, ob2);
      oa3 = WMMA_BF16(a0.v, b3.v, oa3);  ob3 = WMMA_BF16(a1.v, b3.v, ob3);
    }
    __syncthreads();                        // protect P_lds for next chunk
  }

  // ---- final denominator: sum partial s across lanes, then across waves ---
  #pragma unroll
  for (int r = 0; r < 8; ++r) {
    #pragma unroll
    for (int msk = 1; msk < 16; msk <<= 1) {
      s0[r] += __shfl_xor(s0[r], msk, 16);
      s1[r] += __shfl_xor(s1[r], msk, 16);
    }
  }
  if (l16 == 0) {
    #pragma unroll
    for (int r = 0; r < 8; ++r) {
      redA[w][r + 8 * hi]      = s0[r];
      redA[w][16 + r + 8 * hi] = s1[r];
    }
  }
  __syncthreads();
  if (tid < MT) {
    float S = 0.f;
    #pragma unroll
    for (int wv = 0; wv < 8; ++wv) S += redA[wv][tid];
    redB[tid] = S;
  }
  __syncthreads();
  float fsA[8], fsB[8];
  #pragma unroll
  for (int r = 0; r < 8; ++r) {
    fsA[r] = redB[r + 8 * hi];
    fsB[r] = redB[16 + r + 8 * hi];
  }

  // ---- epilogue: /= sumexp, + lang_emb, special-token row override --------
  const float* lf = langF + (size_t)mbase * DIM;
  #pragma unroll
  for (int nt = 0; nt < 4; ++nt) {
    v8f oA = (nt == 0 ? oa0 : nt == 1 ? oa1 : nt == 2 ? oa2 : oa3);
    v8f oB = (nt == 0 ? ob0 : nt == 1 ? ob1 : nt == 2 ? ob2 : ob3);
    int d = dbase + nt * 16 + l16;
    #pragma unroll
    for (int r = 0; r < 8; ++r) {
      int rowA = r + 8 * hi, rowB = 16 + r + 8 * hi;
      float vA = oA[r] / fsA[r] + lf[(size_t)rowA * DIM + d];
      float vB = oB[r] / fsB[r] + lf[(size_t)rowB * DIM + d];
      if ((unsigned)firA[r] < 4u) vA = ctab[(size_t)firA[r] * DIM + d];
      if ((unsigned)firB[r] < 4u) vB = ctab[(size_t)firB[r] * DIM + d];
      out[((size_t)(mbase + rowA)) * DIM + d] = vA;
      out[((size_t)(mbase + rowB)) * DIM + d] = vB;
    }
  }
}

// ---------------------------------------------------------------------------
extern "C" void kernel_launch(void* const* d_in, const int* in_sizes, int n_in,
                              void* d_out, int out_size, void* d_ws, size_t ws_size,
                              hipStream_t stream) {
  (void)in_sizes; (void)n_in; (void)out_size; (void)ws_size;
  const int*   x      = (const int*)d_in[0];
  const float* ctab   = (const float*)d_in[1];
  const float* latent = (const float*)d_in[2];

  char* ws = (char*)d_ws;
  float*          langF = (float*)(ws);                              // 16.78 MB
  unsigned short* langB = (unsigned short*)(ws + 16777216);          //  8.39 MB
  unsigned short* latB  = (unsigned short*)(ws + 25165824);          // 10.24 MB
  unsigned short* latT  = (unsigned short*)(ws + 35405824);          // 10.35 MB

  k_convert<<<(LATENT * DIM + 255) / 256, 256, 0, stream>>>(latent, latB);
  dim3 tg(LT_STRIDE / TT, DIM / TT);        // 158 x 8 tiles, covers zero tail
  k_transpose<<<tg, 256, 0, stream>>>(latent, latT);
  k_lang<<<ROWS, 256, 0, stream>>>(x, ctab, langF, langB);
  k_sde<<<ROWS / MT, 256, 0, stream>>>(x, ctab, langF, langB, latB, latT,
                                       (float*)d_out);
}
